// MultiwayTransformerAttention_79998060855667
// MI455X (gfx1250) — compile-verified, matrix operand order checked by reading, not run on hardware
//
#include <hip/hip_runtime.h>
#include <math.h>

// ---------------------------------------------------------------------------
// MultiwayTransformerAttention for MI455X (gfx1250, wave32, WMMA bf16)
//   B=2, T=2048, D=768, H=12, HD=64
// ---------------------------------------------------------------------------

#define Bb 2
#define Tt 2048
#define Dd 768
#define Hh 12
#define HD 64

typedef __attribute__((ext_vector_type(16))) __bf16 v16bf;
typedef __attribute__((ext_vector_type(8)))  float  v8f;
typedef __attribute__((ext_vector_type(8)))  unsigned short ushortx8;

union FragBF {
    v16bf v;
    unsigned short us[16];
    ushortx8 u8[2];
};

// Manual RNE f32->bf16 (used in one-time conversion kernels)
__device__ __forceinline__ unsigned short f2bf(float f) {
    unsigned int u = __float_as_uint(f);
    unsigned int r = u + 0x7FFFu + ((u >> 16) & 1u);
    return (unsigned short)(r >> 16);
}

// Hardware packed converter (1 VALU op) — used in hot paths
__device__ __forceinline__ unsigned short f2bf_fast(float f) {
    unsigned int pk;
    __asm__ volatile("v_cvt_pk_bf16_f32 %0, %1, %2" : "=v"(pk) : "v"(f), "v"(f));
    return (unsigned short)(pk & 0xFFFFu);
}

// ---------------------------------------------------------------------------
// Kernel 0: f32 -> bf16 (raw ushort) conversion (weights + hidden states)
// ---------------------------------------------------------------------------
__global__ void cvt_bf16_kernel(const float* __restrict__ src,
                                unsigned short* __restrict__ dst, int n) {
    int i = blockIdx.x * blockDim.x + threadIdx.x;
    if (i < n) dst[i] = f2bf(src[i]);
}

// ---------------------------------------------------------------------------
// Kernel 1: fused QKV projection, one wave computes a 16x64 output tile
// (one A fragment reused across 4 B fragments per K-step; 24 K-steps).
//   q,k written as [B,H,T,HD] bf16 ; v written transposed [B,H,HD,T] bf16
// ---------------------------------------------------------------------------
__global__ __launch_bounds__(128)
void qkv_proj_kernel(const unsigned short* __restrict__ hidb,  // [M,768] bf16
                     const unsigned short* __restrict__ wq,
                     const unsigned short* __restrict__ wk,
                     const unsigned short* __restrict__ wv,
                     const float* __restrict__ bq,
                     const float* __restrict__ bk,
                     const float* __restrict__ bv,
                     unsigned short* __restrict__ qo,
                     unsigned short* __restrict__ ko,
                     unsigned short* __restrict__ vt) {
    const int warp = threadIdx.x >> 5;
    const int lane = threadIdx.x & 31;
    const int rl   = lane & 15;
    const int half = lane >> 4;

    const int tile  = blockIdx.x * 4 + warp;          // 9216 tiles total
    const int which = tile / (256 * 12);              // 0=q 1=k 2=v
    const int rem   = tile % (256 * 12);
    const int mt    = rem / 12;
    const int nt    = rem % 12;

    const unsigned short* w = (which == 0) ? wq : (which == 1) ? wk : wv;
    const float*        bia = (which == 0) ? bq : (which == 1) ? bk : bv;

    const int row0 = mt * 16;
    const int col0 = nt * 64;                         // one full head

    v8f c0 = {}, c1 = {}, c2 = {}, c3 = {};
    const unsigned short* arow = hidb + (size_t)(row0 + rl) * Dd;
    const unsigned short* w0 = w + (size_t)(col0 + rl) * Dd;
    const unsigned short* w1 = w0 + (size_t)16 * Dd;
    const unsigned short* w2 = w1 + (size_t)16 * Dd;
    const unsigned short* w3 = w2 + (size_t)16 * Dd;

    for (int k0 = 0; k0 < Dd; k0 += 32) {
        FragBF a, b0, b1, b2, b3;
        a.u8[0] = *(const ushortx8*)(arow + k0 + half * 8);
        a.u8[1] = *(const ushortx8*)(arow + k0 + 16 + half * 8);
        b0.u8[0] = *(const ushortx8*)(w0 + k0 + half * 8);
        b0.u8[1] = *(const ushortx8*)(w0 + k0 + 16 + half * 8);
        b1.u8[0] = *(const ushortx8*)(w1 + k0 + half * 8);
        b1.u8[1] = *(const ushortx8*)(w1 + k0 + 16 + half * 8);
        b2.u8[0] = *(const ushortx8*)(w2 + k0 + half * 8);
        b2.u8[1] = *(const ushortx8*)(w2 + k0 + 16 + half * 8);
        b3.u8[0] = *(const ushortx8*)(w3 + k0 + half * 8);
        b3.u8[1] = *(const ushortx8*)(w3 + k0 + 16 + half * 8);
        c0 = __builtin_amdgcn_wmma_f32_16x16x32_bf16(false, a.v, false, b0.v,
                                                     (short)0, c0, false, false);
        c1 = __builtin_amdgcn_wmma_f32_16x16x32_bf16(false, a.v, false, b1.v,
                                                     (short)0, c1, false, false);
        c2 = __builtin_amdgcn_wmma_f32_16x16x32_bf16(false, a.v, false, b2.v,
                                                     (short)0, c2, false, false);
        c3 = __builtin_amdgcn_wmma_f32_16x16x32_bf16(false, a.v, false, b3.v,
                                                     (short)0, c3, false, false);
    }

    const int   h    = col0 >> 6;
    const float qscl = (which == 0) ? 0.125f : 1.0f;
    const float bn0 = bia[col0 + rl];
    const float bn1 = bia[col0 + 16 + rl];
    const float bn2 = bia[col0 + 32 + rl];
    const float bn3 = bia[col0 + 48 + rl];

    #pragma unroll
    for (int i = 0; i < 8; ++i) {
        const int m  = i + 8 * half;
        const int gm = row0 + m;
        const int b  = gm >> 11;              // /T
        const int t  = gm & (Tt - 1);
        float v0 = (c0[i] + bn0) * qscl;
        float v1 = (c1[i] + bn1) * qscl;
        float v2 = (c2[i] + bn2) * qscl;
        float v3 = (c3[i] + bn3) * qscl;
        if (which == 2) {
            unsigned short* vb = vt + (((size_t)b * Hh + h) * HD) * Tt + t;
            vb[(size_t)(rl)      * Tt] = f2bf_fast(v0);
            vb[(size_t)(16 + rl) * Tt] = f2bf_fast(v1);
            vb[(size_t)(32 + rl) * Tt] = f2bf_fast(v2);
            vb[(size_t)(48 + rl) * Tt] = f2bf_fast(v3);
        } else {
            unsigned short* dst = ((which == 0) ? qo : ko)
                                  + (((size_t)b * Hh + h) * Tt + t) * HD;
            dst[rl]      = f2bf_fast(v0);
            dst[16 + rl] = f2bf_fast(v1);
            dst[32 + rl] = f2bf_fast(v2);
            dst[48 + rl] = f2bf_fast(v3);
        }
    }
}

// ---------------------------------------------------------------------------
// Kernel 2: flash attention. One wave owns a 16-query tile for one (b,h).
// Streams 64 key blocks of 32, online softmax, accumulates 16x64 context.
// ---------------------------------------------------------------------------
__global__ __launch_bounds__(128)
void attn_kernel(const unsigned short* __restrict__ q,   // [B,H,T,HD]
                 const unsigned short* __restrict__ k,   // [B,H,T,HD]
                 const unsigned short* __restrict__ vt,  // [B,H,HD,T]
                 const float* __restrict__ bias,         // [H,T,T]
                 const float* __restrict__ mask,         // [B,1,T,T]
                 unsigned short* __restrict__ ctx) {     // [B,T,D] bf16
    __shared__ __align__(16) unsigned short pT[4][16 * 32];

    const int warp = threadIdx.x >> 5;
    const int lane = threadIdx.x & 31;
    const int rl   = lane & 15;
    const int half = lane >> 4;

    const int tid = blockIdx.x * 4 + warp;   // 3072 tiles = 24 bh * 128 ttiles
    const int bh  = tid >> 7;
    const int tt  = tid & 127;
    const int b   = bh / Hh;
    const int h   = bh % Hh;
    const int t0  = tt * 16;

    // Q A-fragments (K = HD = 64 -> two 16x32 fragments), kept in registers
    FragBF aQ0, aQ1;
    {
        const unsigned short* qrow = q + ((size_t)bh * Tt + t0 + rl) * HD;
        aQ0.u8[0] = *(const ushortx8*)(qrow + half * 8);
        aQ0.u8[1] = *(const ushortx8*)(qrow + 16 + half * 8);
        aQ1.u8[0] = *(const ushortx8*)(qrow + 32 + half * 8);
        aQ1.u8[1] = *(const ushortx8*)(qrow + 32 + 16 + half * 8);
    }

    v8f acc0 = {}, acc1 = {}, acc2 = {}, acc3 = {};
    float mrow[8], lrow[8];
    #pragma unroll
    for (int i = 0; i < 8; ++i) { mrow[i] = -1e30f; lrow[i] = 0.0f; }

    const float* biasH = bias + (size_t)h * Tt * Tt;
    const float* maskB = mask + (size_t)b * Tt * Tt;
    unsigned short* pW = &pT[warp][0];

    for (int s0 = 0; s0 < Tt; s0 += 32) {
        // ---- scores: 16x32 tile = two 16x16 WMMA accumulators, K = 64 ----
        v8f s0v = {}, s1v = {};
        {
            const unsigned short* kr0 = k + ((size_t)bh * Tt + s0 + rl) * HD;
            const unsigned short* kr1 = k + ((size_t)bh * Tt + s0 + 16 + rl) * HD;
            FragBF bk0, bk1;
            bk0.u8[0] = *(const ushortx8*)(kr0 + half * 8);
            bk0.u8[1] = *(const ushortx8*)(kr0 + 16 + half * 8);
            bk1.u8[0] = *(const ushortx8*)(kr1 + half * 8);
            bk1.u8[1] = *(const ushortx8*)(kr1 + 16 + half * 8);
            s0v = __builtin_amdgcn_wmma_f32_16x16x32_bf16(false, aQ0.v, false, bk0.v,
                                                          (short)0, s0v, false, false);
            s1v = __builtin_amdgcn_wmma_f32_16x16x32_bf16(false, aQ0.v, false, bk1.v,
                                                          (short)0, s1v, false, false);
            bk0.u8[0] = *(const ushortx8*)(kr0 + 32 + half * 8);
            bk0.u8[1] = *(const ushortx8*)(kr0 + 32 + 16 + half * 8);
            bk1.u8[0] = *(const ushortx8*)(kr1 + 32 + half * 8);
            bk1.u8[1] = *(const ushortx8*)(kr1 + 32 + 16 + half * 8);
            s0v = __builtin_amdgcn_wmma_f32_16x16x32_bf16(false, aQ1.v, false, bk0.v,
                                                          (short)0, s0v, false, false);
            s1v = __builtin_amdgcn_wmma_f32_16x16x32_bf16(false, aQ1.v, false, bk1.v,
                                                          (short)0, s1v, false, false);
        }

        // ---- add rel-pos bias + mask, online softmax ----
        float p0[8], p1[8];
        #pragma unroll
        for (int i = 0; i < 8; ++i) {
            const int m  = i + 8 * half;
            const size_t roff = (size_t)(t0 + m) * Tt;
            const int sa = s0 + rl;
            const int sb = s0 + 16 + rl;
            float v0 = s0v[i] + biasH[roff + sa] + maskB[roff + sa];
            float v1 = s1v[i] + biasH[roff + sb] + maskB[roff + sb];

            float rm = fmaxf(v0, v1);
            rm = fmaxf(rm, __shfl_xor(rm, 1, 32));
            rm = fmaxf(rm, __shfl_xor(rm, 2, 32));
            rm = fmaxf(rm, __shfl_xor(rm, 4, 32));
            rm = fmaxf(rm, __shfl_xor(rm, 8, 32));

            float mnew  = fmaxf(mrow[i], rm);
            float alpha = __expf(mrow[i] - mnew);
            float e0 = __expf(v0 - mnew);
            float e1 = __expf(v1 - mnew);
            float rs = e0 + e1;
            rs += __shfl_xor(rs, 1, 32);
            rs += __shfl_xor(rs, 2, 32);
            rs += __shfl_xor(rs, 4, 32);
            rs += __shfl_xor(rs, 8, 32);

            lrow[i] = lrow[i] * alpha + rs;
            mrow[i] = mnew;
            acc0[i] *= alpha; acc1[i] *= alpha;
            acc2[i] *= alpha; acc3[i] *= alpha;
            p0[i] = e0; p1[i] = e1;
        }

        // ---- stage P (bf16) in wave-private LDS, reload in A-frag layout ----
        __asm__ volatile("" ::: "memory");
        #pragma unroll
        for (int i = 0; i < 8; ++i) {
            const int m = i + 8 * half;
            pW[m * 32 + rl]      = f2bf_fast(p0[i]);
            pW[m * 32 + 16 + rl] = f2bf_fast(p1[i]);
        }
        __asm__ volatile("s_wait_dscnt 0" ::: "memory");

        FragBF aP;
        aP.u8[0] = *(const ushortx8*)(&pW[rl * 32 + half * 8]);
        aP.u8[1] = *(const ushortx8*)(&pW[rl * 32 + 16 + half * 8]);

        // ---- acc += P @ V : 4 hd-tiles of 16, K = 32 (V stored transposed) ----
        {
            const unsigned short* vb = vt + ((size_t)bh * HD + rl) * Tt + s0;
            FragBF bv;
            bv.u8[0] = *(const ushortx8*)(vb + half * 8);
            bv.u8[1] = *(const ushortx8*)(vb + 16 + half * 8);
            acc0 = __builtin_amdgcn_wmma_f32_16x16x32_bf16(false, aP.v, false, bv.v,
                                                           (short)0, acc0, false, false);
            vb += (size_t)16 * Tt;
            bv.u8[0] = *(const ushortx8*)(vb + half * 8);
            bv.u8[1] = *(const ushortx8*)(vb + 16 + half * 8);
            acc1 = __builtin_amdgcn_wmma_f32_16x16x32_bf16(false, aP.v, false, bv.v,
                                                           (short)0, acc1, false, false);
            vb += (size_t)16 * Tt;
            bv.u8[0] = *(const ushortx8*)(vb + half * 8);
            bv.u8[1] = *(const ushortx8*)(vb + 16 + half * 8);
            acc2 = __builtin_amdgcn_wmma_f32_16x16x32_bf16(false, aP.v, false, bv.v,
                                                           (short)0, acc2, false, false);
            vb += (size_t)16 * Tt;
            bv.u8[0] = *(const ushortx8*)(vb + half * 8);
            bv.u8[1] = *(const ushortx8*)(vb + 16 + half * 8);
            acc3 = __builtin_amdgcn_wmma_f32_16x16x32_bf16(false, aP.v, false, bv.v,
                                                           (short)0, acc3, false, false);
        }
    }

    // ---- normalize and store context tile (bf16, [B,T,D]) ----
    #pragma unroll
    for (int i = 0; i < 8; ++i) {
        const int m = i + 8 * half;
        const float inv = 1.0f / lrow[i];
        unsigned short* crow = ctx + ((size_t)b * Tt + t0 + m) * Dd + h * HD;
        crow[rl]      = f2bf_fast(acc0[i] * inv);
        crow[16 + rl] = f2bf_fast(acc1[i] * inv);
        crow[32 + rl] = f2bf_fast(acc2[i] * inv);
        crow[48 + rl] = f2bf_fast(acc3[i] * inv);
    }
}

// ---------------------------------------------------------------------------
// Kernel 3: output projection: out = ctx @ Wo^T + bo (f32 out), 16x64 tiles
// ---------------------------------------------------------------------------
__global__ __launch_bounds__(128)
void out_proj_kernel(const unsigned short* __restrict__ ctx,  // [M,768] bf16
                     const unsigned short* __restrict__ wo,   // [768,768] bf16
                     const float* __restrict__ bo,
                     float* __restrict__ out) {               // [M,768] f32
    const int warp = threadIdx.x >> 5;
    const int lane = threadIdx.x & 31;
    const int rl   = lane & 15;
    const int half = lane >> 4;

    const int tile = blockIdx.x * 4 + warp;   // 3072 tiles (256 x 12)
    const int mt = tile / 12;
    const int nt = tile % 12;
    const int row0 = mt * 16;
    const int col0 = nt * 64;

    v8f c0 = {}, c1 = {}, c2 = {}, c3 = {};
    const unsigned short* arow = ctx + (size_t)(row0 + rl) * Dd;
    const unsigned short* w0 = wo + (size_t)(col0 + rl) * Dd;
    const unsigned short* w1 = w0 + (size_t)16 * Dd;
    const unsigned short* w2 = w1 + (size_t)16 * Dd;
    const unsigned short* w3 = w2 + (size_t)16 * Dd;

    for (int k0 = 0; k0 < Dd; k0 += 32) {
        FragBF a, b0, b1, b2, b3;
        a.u8[0] = *(const ushortx8*)(arow + k0 + half * 8);
        a.u8[1] = *(const ushortx8*)(arow + k0 + 16 + half * 8);
        b0.u8[0] = *(const ushortx8*)(w0 + k0 + half * 8);
        b0.u8[1] = *(const ushortx8*)(w0 + k0 + 16 + half * 8);
        b1.u8[0] = *(const ushortx8*)(w1 + k0 + half * 8);
        b1.u8[1] = *(const ushortx8*)(w1 + k0 + 16 + half * 8);
        b2.u8[0] = *(const ushortx8*)(w2 + k0 + half * 8);
        b2.u8[1] = *(const ushortx8*)(w2 + k0 + 16 + half * 8);
        b3.u8[0] = *(const ushortx8*)(w3 + k0 + half * 8);
        b3.u8[1] = *(const ushortx8*)(w3 + k0 + 16 + half * 8);
        c0 = __builtin_amdgcn_wmma_f32_16x16x32_bf16(false, a.v, false, b0.v,
                                                     (short)0, c0, false, false);
        c1 = __builtin_amdgcn_wmma_f32_16x16x32_bf16(false, a.v, false, b1.v,
                                                     (short)0, c1, false, false);
        c2 = __builtin_amdgcn_wmma_f32_16x16x32_bf16(false, a.v, false, b2.v,
                                                     (short)0, c2, false, false);
        c3 = __builtin_amdgcn_wmma_f32_16x16x32_bf16(false, a.v, false, b3.v,
                                                     (short)0, c3, false, false);
    }

    const float bn0 = bo[col0 + rl];
    const float bn1 = bo[col0 + 16 + rl];
    const float bn2 = bo[col0 + 32 + rl];
    const float bn3 = bo[col0 + 48 + rl];

    #pragma unroll
    for (int i = 0; i < 8; ++i) {
        const int m = i + 8 * half;
        float* orow = out + (size_t)(row0 + m) * Dd + col0;
        orow[rl]      = c0[i] + bn0;
        orow[16 + rl] = c1[i] + bn1;
        orow[32 + rl] = c2[i] + bn2;
        orow[48 + rl] = c3[i] + bn3;
    }
}

// ---------------------------------------------------------------------------
extern "C" void kernel_launch(void* const* d_in, const int* in_sizes, int n_in,
                              void* d_out, int out_size, void* d_ws, size_t ws_size,
                              hipStream_t stream) {
    const float* hid  = (const float*)d_in[0];
    const float* mask = (const float*)d_in[1];
    const float* bias = (const float*)d_in[2];
    const float* Wq = (const float*)d_in[3];
    const float* bq = (const float*)d_in[4];
    const float* Wk = (const float*)d_in[5];
    const float* bk = (const float*)d_in[6];
    const float* Wv = (const float*)d_in[7];
    const float* bv = (const float*)d_in[8];
    const float* Wo = (const float*)d_in[9];
    const float* bo = (const float*)d_in[10];
    float* out = (float*)d_out;

    const size_t WN = (size_t)Dd * Dd;            // 589824
    const size_t QN = (size_t)Bb * Hh * Tt * HD;  // 3145728 (== B*T*D)

    unsigned short* ws   = (unsigned short*)d_ws;
    unsigned short* wqb  = ws;
    unsigned short* wkb  = ws + WN;
    unsigned short* wvb  = ws + 2 * WN;
    unsigned short* wob  = ws + 3 * WN;
    unsigned short* qb   = ws + 4 * WN;
    unsigned short* kb   = qb + QN;
    unsigned short* vtb  = kb + QN;
    unsigned short* ctx  = vtb + QN;
    unsigned short* hidb = ctx + QN;

    // 0) one-time f32 -> bf16 conversions
    const int cvtW = (int)((WN + 255) / 256);
    const int cvtH = (int)((QN + 255) / 256);
    cvt_bf16_kernel<<<cvtW, 256, 0, stream>>>(Wq, wqb, (int)WN);
    cvt_bf16_kernel<<<cvtW, 256, 0, stream>>>(Wk, wkb, (int)WN);
    cvt_bf16_kernel<<<cvtW, 256, 0, stream>>>(Wv, wvb, (int)WN);
    cvt_bf16_kernel<<<cvtW, 256, 0, stream>>>(Wo, wob, (int)WN);
    cvt_bf16_kernel<<<cvtH, 256, 0, stream>>>(hid, hidb, (int)QN);

    // 1) fused QKV projection (9216 tiles of 16x64 / 4 waves per block)
    qkv_proj_kernel<<<2304, 128, 0, stream>>>(hidb, wqb, wkb, wvb, bq, bk, bv,
                                              qb, kb, vtb);

    // 2) flash attention (3072 tiles / 4 waves per block)
    attn_kernel<<<768, 128, 0, stream>>>(qb, kb, vtb, bias, mask, ctx);

    // 3) output projection (3072 tiles of 16x64 / 4 waves per block)
    out_proj_kernel<<<768, 128, 0, stream>>>(ctx, wob, bo, out);
}
